// GCNWithGAT_7713761263782
// MI455X (gfx1250) — matrix operand order repaired and checked by reading, not compiled, not run on hardware
//
#include <hip/hip_runtime.h>

// ---------------- problem constants ----------------
constexpr int BB = 16;      // batch
constexpr int NN = 2048;    // nodes
constexpr int NF = 128;     // nfeat
constexpr int NH = 128;     // nhid
constexpr int NO = 64;      // nout
constexpr int GROWS = BB * NN;           // 32768 flattened rows
constexpr float BN_EPS = 1e-5f;
constexpr float LRELU = 0.2f;

// ---------------- WMMA types ----------------
typedef _Float16 h16;
typedef __attribute__((ext_vector_type(16))) _Float16 v16h;
typedef __attribute__((ext_vector_type(8)))  _Float16 v8h;
typedef __attribute__((ext_vector_type(8)))  float    v8f;

#define WMMA_F16(a, b, c) \
  __builtin_amdgcn_wmma_f32_16x16x32_f16(false, (a), false, (b), (short)0, (c), false, false)

// A-fragment: 16x32 f16, row-major source, leading dim ld (elements).
// Lane L: row = row0 + L%16; elems 0..7 <- k0+hi*8+[0..8), elems 8..15 <- k0+16+hi*8+[0..8)
__device__ __forceinline__ v16h load_a_frag(const h16* __restrict__ base, int ld,
                                            int row0, int k0) {
  int lane = threadIdx.x & 31;
  int hi = lane >> 4;
  const h16* p = base + (size_t)(row0 + (lane & 15)) * ld + k0 + hi * 8;
  v8h lo = *(const v8h*)p;
  v8h hiu = *(const v8h*)(p + 16);
  v16h a;
#pragma unroll
  for (int i = 0; i < 8; ++i) { a[i] = lo[i]; a[i + 8] = hiu[i]; }
  return a;
}

// B-fragment: 32x16 f16 where B[k][j] = src[(col0+j)*ld + k]  (k contiguous per lane).
// Lane L: col = col0 + L%16; elems 0..7 <- k0+hi*8+[0..8), elems 8..15 <- k0+16+hi*8+[0..8)
__device__ __forceinline__ v16h load_b_frag(const h16* __restrict__ base, int ld,
                                            int col0, int k0) {
  int lane = threadIdx.x & 31;
  int hi = lane >> 4;
  const h16* p = base + (size_t)(col0 + (lane & 15)) * ld + k0 + hi * 8;
  v8h lo = *(const v8h*)p;
  v8h hiu = *(const v8h*)(p + 16);
  v16h b;
#pragma unroll
  for (int i = 0; i < 8; ++i) { b[i] = lo[i]; b[i + 8] = hiu[i]; }
  return b;
}

// ---------------- prep kernels ----------------
__global__ void cvt_f16(const float* __restrict__ in, h16* __restrict__ out, int n) {
  int i = blockIdx.x * blockDim.x + threadIdx.x;
  if (i < n) out[i] = (h16)in[i];
}

// in[rows][cols] (f32) -> out[cols][rows] (f16)
__global__ void transpose_cvt(const float* __restrict__ in, h16* __restrict__ out,
                              int rows, int cols) {
  int i = blockIdx.x * blockDim.x + threadIdx.x;
  if (i < rows * cols) {
    int r = i / cols, c = i % cols;
    out[(size_t)c * rows + r] = (h16)in[i];
  }
}

// ---------------- GEMM 1: support = x @ W, stored transposed f16 ----------------
// Wave computes 16 rows x all 128 cols (8 col tiles), A-frag reused 8x per K chunk.
__global__ void gemm_support(const h16* __restrict__ xh, const h16* __restrict__ WT,
                             h16* __restrict__ supT) {
  int wid = (blockIdx.x * blockDim.x + threadIdx.x) >> 5;   // 0..2047
  int row0 = wid * 16;
  constexpr int CT = NH / 16;                               // 8
  v8f zero = {};
  v8f acc[CT];
#pragma unroll
  for (int t = 0; t < CT; ++t) acc[t] = zero;
#pragma unroll
  for (int kc = 0; kc < NF; kc += 32) {
    v16h a = load_a_frag(xh, NF, row0, kc);
#pragma unroll
    for (int t = 0; t < CT; ++t) {
      v16h b = load_b_frag(WT, NF, t * 16, kc);
      acc[t] = WMMA_F16(a, b, acc[t]);
    }
  }
  int lane = threadIdx.x & 31;
  int j = lane & 15, hi = lane >> 4;
  int g0 = row0 + hi * 8;               // 8 consecutive global rows (same batch)
  int bat = g0 / NN, n0 = g0 % NN;
#pragma unroll
  for (int t = 0; t < CT; ++t) {
    v8h o;
#pragma unroll
    for (int r = 0; r < 8; ++r) o[r] = (h16)acc[t][r];
    *(v8h*)(supT + ((size_t)(bat * NH + t * 16 + j)) * NN + n0) = o;
  }
}

// ---------------- GEMM 2: h = adj @ support + bias (f32 out) ----------------
// Wave computes 32 rows x all 128 cols: 2 A-frags + 8 B-frags -> 16 WMMA per K chunk.
__global__ void gemm_adj(const h16* __restrict__ adjh, const h16* __restrict__ supT,
                         const float* __restrict__ bias, float* __restrict__ h) {
  int wid = (blockIdx.x * blockDim.x + threadIdx.x) >> 5;   // 0..1023
  constexpr int RB = NN / 32;   // 64 row-blocks per batch
  constexpr int CT = NH / 16;   // 8 col tiles
  int bat = wid / RB;
  int row0 = (wid % RB) * 32;
  const h16* supTb = supT + (size_t)bat * NH * NN;
  v8f zero = {};
  v8f acc[2][CT];
#pragma unroll
  for (int rf = 0; rf < 2; ++rf)
#pragma unroll
    for (int t = 0; t < CT; ++t) acc[rf][t] = zero;
#pragma unroll 2
  for (int kc = 0; kc < NN; kc += 32) {
    v16h a0 = load_a_frag(adjh, NN, row0, kc);
    v16h a1 = load_a_frag(adjh, NN, row0 + 16, kc);
#pragma unroll
    for (int t = 0; t < CT; ++t) {
      v16h b = load_b_frag(supTb, NN, t * 16, kc);
      acc[0][t] = WMMA_F16(a0, b, acc[0][t]);
      acc[1][t] = WMMA_F16(a1, b, acc[1][t]);
    }
  }
  int lane = threadIdx.x & 31;
  int j = lane & 15, hi = lane >> 4;
#pragma unroll
  for (int rf = 0; rf < 2; ++rf) {
    float* outp = h + ((size_t)bat * NN + row0 + rf * 16 + hi * 8) * NH + j;
#pragma unroll
    for (int t = 0; t < CT; ++t) {
      float bv = bias[t * 16 + j];
#pragma unroll
      for (int r = 0; r < 8; ++r) outp[(size_t)r * NH + t * 16] = acc[rf][t][r] + bv;
    }
  }
}

// ---------------- BatchNorm ----------------
__global__ void bn_zero(float* sums, float* sumsq) {
  int i = threadIdx.x;
  if (i < NH) { sums[i] = 0.0f; sumsq[i] = 0.0f; }
}

__global__ void bn_stats(const float* __restrict__ h, float* sums, float* sumsq) {
  int c = threadIdx.x;                       // 128 threads = channels
  size_t row0 = (size_t)blockIdx.x * 128;    // 128 rows per block
  float s = 0.0f, q = 0.0f;
  for (int r = 0; r < 128; ++r) {
    float v = h[(row0 + r) * NH + c];
    s += v; q += v * v;
  }
  atomicAdd(&sums[c], s);
  atomicAdd(&sumsq[c], q);
}

__global__ void bn_finalize(const float* sums, const float* sumsq,
                            const float* __restrict__ gamma, const float* __restrict__ beta,
                            float* scale, float* shift) {
  int c = threadIdx.x;
  if (c < NH) {
    constexpr float inv = 1.0f / (float)GROWS;
    float mean = sums[c] * inv;
    float var  = sumsq[c] * inv - mean * mean;
    float sc = gamma[c] * rsqrtf(var + BN_EPS);
    scale[c] = sc;
    shift[c] = beta[c] - mean * sc;
  }
}

__global__ void bn_apply_relu(const float* __restrict__ h, const float* __restrict__ scale,
                              const float* __restrict__ shift, h16* __restrict__ hn, int n) {
  int i = blockIdx.x * blockDim.x + threadIdx.x;
  if (i < n) {
    int c = i & (NH - 1);
    float v = h[i] * scale[c] + shift[c];
    hn[i] = (h16)(v > 0.0f ? v : 0.0f);
  }
}

// ---------------- GEMM 3: hp = hn @ gat_W (f32 + f16 out) ----------------
// Wave computes 16 rows x all 64 cols (4 col tiles).
__global__ void gemm_hp(const h16* __restrict__ hn, const h16* __restrict__ gatWT,
                        float* __restrict__ hpf, h16* __restrict__ hph) {
  int wid = (blockIdx.x * blockDim.x + threadIdx.x) >> 5;   // 0..2047
  int row0 = wid * 16;
  constexpr int CT = NO / 16;      // 4
  v8f zero = {};
  v8f acc[CT];
#pragma unroll
  for (int t = 0; t < CT; ++t) acc[t] = zero;
#pragma unroll
  for (int kc = 0; kc < NH; kc += 32) {
    v16h a = load_a_frag(hn, NH, row0, kc);
#pragma unroll
    for (int t = 0; t < CT; ++t) {
      v16h b = load_b_frag(gatWT, NH, t * 16, kc);
      acc[t] = WMMA_F16(a, b, acc[t]);
    }
  }
  int lane = threadIdx.x & 31;
  int j = lane & 15, hi = lane >> 4;
  size_t base = (size_t)(row0 + hi * 8) * NO + j;
#pragma unroll
  for (int t = 0; t < CT; ++t) {
#pragma unroll
    for (int r = 0; r < 8; ++r) {
      size_t idx = base + (size_t)r * NO + t * 16;
      hpf[idx] = acc[t][r];
      hph[idx] = (h16)acc[t][r];
    }
  }
}

// ---------------- Attention pass 1: column softmax stats over n ----------------
// Wave owns 4 m-tiles (64 cols); B-frags stay resident, A-frags stream over n.
__global__ void attn_pass1(const h16* __restrict__ hph,
                           float* __restrict__ colmax, float* __restrict__ colsum) {
  constexpr int MT = 4;
  int wave = threadIdx.x >> 5;
  int mg = blockIdx.x * (blockDim.x >> 5) + wave;  // 0..31
  int bat = blockIdx.y;
  int m0 = mg * (MT * 16);
  const h16* hp = hph + (size_t)bat * NN * NO;
  v16h bf[MT][2];
#pragma unroll
  for (int t = 0; t < MT; ++t) {
    bf[t][0] = load_b_frag(hp, NO, m0 + t * 16, 0);
    bf[t][1] = load_b_frag(hp, NO, m0 + t * 16, 32);
  }
  float mx[MT], sm[MT];
#pragma unroll
  for (int t = 0; t < MT; ++t) { mx[t] = -3.0e38f; sm[t] = 0.0f; }
  for (int n0 = 0; n0 < NN; n0 += 16) {
    v16h a0 = load_a_frag(hp, NO, n0, 0);
    v16h a1 = load_a_frag(hp, NO, n0, 32);
#pragma unroll
    for (int t = 0; t < MT; ++t) {
      v8f c = {};
      c = WMMA_F16(a0, bf[t][0], c);
      c = WMMA_F16(a1, bf[t][1], c);
#pragma unroll
      for (int r = 0; r < 8; ++r) {
        float e = c[r];
        e = e > 0.0f ? e : LRELU * e;
        if (e > mx[t]) { sm[t] = sm[t] * __expf(mx[t] - e) + 1.0f; mx[t] = e; }
        else           { sm[t] += __expf(e - mx[t]); }
      }
    }
  }
  int lane = threadIdx.x & 31;
#pragma unroll
  for (int t = 0; t < MT; ++t) {
    float omx = __shfl_xor(mx[t], 16);
    float osm = __shfl_xor(sm[t], 16);
    float nm = fmaxf(mx[t], omx);
    float s  = sm[t] * __expf(mx[t] - nm) + osm * __expf(omx - nm);
    if (lane < 16) {
      colmax[(size_t)bat * NN + m0 + t * 16 + lane] = nm;
      colsum[(size_t)bat * NN + m0 + t * 16 + lane] = s;
    }
  }
}

// ---------------- Attention pass 2: S[b,n] = sum_m attention[b,n,m] ----------------
// Wave owns 4 n-tiles (64 rows); A-frags stay resident, B-frags stream over m.
__global__ void attn_pass2(const h16* __restrict__ hph,
                           const float* __restrict__ colmax, const float* __restrict__ colsum,
                           float* __restrict__ S) {
  constexpr int NT = 4;
  int wave = threadIdx.x >> 5;
  int ng = blockIdx.x * (blockDim.x >> 5) + wave;  // 0..31
  int bat = blockIdx.y;
  int n0 = ng * (NT * 16);
  const h16* hp = hph + (size_t)bat * NN * NO;
  v16h af[NT][2];
#pragma unroll
  for (int t = 0; t < NT; ++t) {
    af[t][0] = load_a_frag(hp, NO, n0 + t * 16, 0);
    af[t][1] = load_a_frag(hp, NO, n0 + t * 16, 32);
  }
  int lane = threadIdx.x & 31;
  int j = lane & 15, hi = lane >> 4;
  const float* cmb = colmax + (size_t)bat * NN;
  const float* csb = colsum + (size_t)bat * NN;
  float acc[NT][8];
#pragma unroll
  for (int t = 0; t < NT; ++t)
#pragma unroll
    for (int r = 0; r < 8; ++r) acc[t][r] = 0.0f;
  for (int m0 = 0; m0 < NN; m0 += 16) {
    v16h b0 = load_b_frag(hp, NO, m0, 0);
    v16h b1 = load_b_frag(hp, NO, m0, 32);
    float cm  = cmb[m0 + j];
    float inv = 1.0f / csb[m0 + j];
#pragma unroll
    for (int t = 0; t < NT; ++t) {
      v8f c = {};
      c = WMMA_F16(af[t][0], b0, c);
      c = WMMA_F16(af[t][1], b1, c);
#pragma unroll
      for (int r = 0; r < 8; ++r) {
        float e = c[r];
        e = e > 0.0f ? e : LRELU * e;
        acc[t][r] += __expf(e - cm) * inv;
      }
    }
  }
  // butterfly sum across the 16 columns within each half
#pragma unroll
  for (int mask = 1; mask < 16; mask <<= 1) {
#pragma unroll
    for (int t = 0; t < NT; ++t)
#pragma unroll
      for (int r = 0; r < 8; ++r) acc[t][r] += __shfl_xor(acc[t][r], mask);
  }
  if (j == 0) {
#pragma unroll
    for (int t = 0; t < NT; ++t)
#pragma unroll
      for (int r = 0; r < 8; ++r)
        S[(size_t)bat * NN + n0 + t * 16 + hi * 8 + r] = acc[t][r];
  }
}

// ---------------- final scale: out = hp * S[row] ----------------
__global__ void scale_out(const float* __restrict__ hpf, const float* __restrict__ S,
                          float* __restrict__ out, int total) {
  int i = blockIdx.x * blockDim.x + threadIdx.x;
  if (i < total) {
    int row = i / NO;
    out[i] = hpf[i] * S[row];
  }
}

// ---------------- workspace layout (bytes) ----------------
constexpr size_t SZ_XH   = (size_t)GROWS * NF * 2;     // 8.4 MB
constexpr size_t SZ_ADJH = (size_t)NN * NN * 2;        // 8.4 MB
constexpr size_t SZ_WT   = (size_t)NH * NF * 2;
constexpr size_t SZ_GWT  = (size_t)NO * NH * 2;
constexpr size_t SZ_SUPT = (size_t)BB * NH * NN * 2;   // 8.4 MB
constexpr size_t SZ_H    = (size_t)GROWS * NH * 4;     // 16.8 MB
constexpr size_t SZ_VEC  = (size_t)NH * 4;             // 512 B
constexpr size_t SZ_HN   = (size_t)GROWS * NH * 2;     // 8.4 MB
constexpr size_t SZ_HPF  = (size_t)GROWS * NO * 4;     // 8.4 MB
constexpr size_t SZ_HPH  = (size_t)GROWS * NO * 2;     // 4.2 MB
constexpr size_t SZ_COL  = (size_t)BB * NN * 4;        // 128 KB

constexpr size_t OFF_XH   = 0;
constexpr size_t OFF_ADJH = OFF_XH + SZ_XH;
constexpr size_t OFF_WT   = OFF_ADJH + SZ_ADJH;
constexpr size_t OFF_GWT  = OFF_WT + SZ_WT;
constexpr size_t OFF_SUPT = OFF_GWT + SZ_GWT;
constexpr size_t OFF_H    = OFF_SUPT + SZ_SUPT;
constexpr size_t OFF_SUM  = OFF_H + SZ_H;
constexpr size_t OFF_SUMQ = OFF_SUM + SZ_VEC;
constexpr size_t OFF_SCL  = OFF_SUMQ + SZ_VEC;
constexpr size_t OFF_SHF  = OFF_SCL + SZ_VEC;
constexpr size_t OFF_HN   = OFF_SHF + SZ_VEC;
constexpr size_t OFF_HPF  = OFF_HN + SZ_HN;
constexpr size_t OFF_HPH  = OFF_HPF + SZ_HPF;
constexpr size_t OFF_CMAX = OFF_HPH + SZ_HPH;
constexpr size_t OFF_CSUM = OFF_CMAX + SZ_COL;
constexpr size_t OFF_S    = OFF_CSUM + SZ_COL;

extern "C" void kernel_launch(void* const* d_in, const int* in_sizes, int n_in,
                              void* d_out, int out_size, void* d_ws, size_t ws_size,
                              hipStream_t stream) {
  const float* x       = (const float*)d_in[0];
  // d_in[1] = adj (unused by the reference GAT)
  const float* adj_gcn = (const float*)d_in[2];
  const float* gcn_w   = (const float*)d_in[3];
  const float* gcn_b   = (const float*)d_in[4];
  const float* gamma   = (const float*)d_in[5];
  const float* beta    = (const float*)d_in[6];
  const float* gat_w   = (const float*)d_in[7];
  float* out = (float*)d_out;

  char* ws = (char*)d_ws;
  h16*   xh    = (h16*)(ws + OFF_XH);
  h16*   adjh  = (h16*)(ws + OFF_ADJH);
  h16*   WT    = (h16*)(ws + OFF_WT);
  h16*   GWT   = (h16*)(ws + OFF_GWT);
  h16*   supT  = (h16*)(ws + OFF_SUPT);
  float* h     = (float*)(ws + OFF_H);
  float* sums  = (float*)(ws + OFF_SUM);
  float* sumsq = (float*)(ws + OFF_SUMQ);
  float* scl   = (float*)(ws + OFF_SCL);
  float* shf   = (float*)(ws + OFF_SHF);
  h16*   hn    = (h16*)(ws + OFF_HN);
  float* hpf   = (float*)(ws + OFF_HPF);
  h16*   hph   = (h16*)(ws + OFF_HPH);
  float* cmax  = (float*)(ws + OFF_CMAX);
  float* csum  = (float*)(ws + OFF_CSUM);
  float* S     = (float*)(ws + OFF_S);

  const int nXH  = GROWS * NF;         // 4.19M
  const int nADJ = NN * NN;            // 4.19M
  const int nHP  = GROWS * NO;         // 2.10M

  // --- prep: f16 conversions / transposes ---
  cvt_f16<<<(nXH + 255) / 256, 256, 0, stream>>>(x, xh, nXH);
  cvt_f16<<<(nADJ + 255) / 256, 256, 0, stream>>>(adj_gcn, adjh, nADJ);
  transpose_cvt<<<(NF * NH + 255) / 256, 256, 0, stream>>>(gcn_w, WT, NF, NH);
  transpose_cvt<<<(NF * NO + 255) / 256, 256, 0, stream>>>(gat_w, GWT, NF, NO);

  // --- support = x @ W (stored transposed) ---
  gemm_support<<<(GROWS / 16) / 4, 128, 0, stream>>>(xh, WT, supT);   // 2048 waves

  // --- h = adj @ support + bias ---
  gemm_adj<<<(BB * (NN / 32)) / 4, 128, 0, stream>>>(adjh, supT, gcn_b, h); // 1024 waves

  // --- BatchNorm (training stats) + ReLU -> hn (f16) ---
  bn_zero<<<1, 128, 0, stream>>>(sums, sumsq);
  bn_stats<<<GROWS / 128, 128, 0, stream>>>(h, sums, sumsq);
  bn_finalize<<<1, 128, 0, stream>>>(sums, sumsq, gamma, beta, scl, shf);
  bn_apply_relu<<<(GROWS * NH + 255) / 256, 256, 0, stream>>>(h, scl, shf, hn, GROWS * NH);

  // --- hp = hn @ gat_W ---
  gemm_hp<<<(GROWS / 16) / 4, 128, 0, stream>>>(hn, GWT, hpf, hph);   // 2048 waves

  // --- two-pass column softmax + row-sum of attention ---
  {
    dim3 grid((NN / 64) / 4, BB);           // (8, 16)
    attn_pass1<<<grid, 128, 0, stream>>>(hph, cmax, csum);
    attn_pass2<<<grid, 128, 0, stream>>>(hph, cmax, csum, S);
  }
  // --- out = hp * S ---
  scale_out<<<(nHP + 255) / 256, 256, 0, stream>>>(hpf, S, out, nHP);

  (void)in_sizes; (void)n_in; (void)out_size; (void)ws_size;
}
